// vgae_206158430566
// MI455X (gfx1250) — matrix operand order repaired and checked by reading, not compiled, not run on hardware
//
#include <hip/hip_runtime.h>
#include <hip/hip_bf16.h>

// ---------------------------------------------------------------------------
// MI455X / gfx1250 VGAE loss. Matrix-core bound: 3x (600k x 128 x 128) GEMMs
// via v_wmma_f32_16x16x32_f16 (f16 in, f32 acc). x (51MB) is L2-resident so
// the edge gathers are cheap; weights live in LDS transposed as f16.
// ---------------------------------------------------------------------------

typedef __attribute__((ext_vector_type(16))) _Float16 v16h;
typedef __attribute__((ext_vector_type(8)))  float    v8f;

union AFrag { v16h v; _Float16 h[16]; };
union CFrag { v8f  v; float     f[8]; };

#define DEE 128

__device__ __forceinline__ float sp_clamp(float z) {
  // min(log(1 + e^z), 100), numerically stable. Matches BCE with -100 clamp.
  float sp = fmaxf(z, 0.f) + log1pf(expf(-fabsf(z)));
  return fminf(sp, 100.f);
}

// A fragment: 16x32 f16 tile of relu(x[ri]*x[rj]) per ISA 7.12.2 layout.
// lane M = lane&15 ; low half lanes K-offset 0, high half 8; VGPR4..7 at K+16.
__device__ __forceinline__ void load_a_em(const float* __restrict__ x,
                                          int ri, int rj, int k0, int kaHalf,
                                          AFrag& a) {
#pragma unroll
  for (int v = 0; v < 8; ++v) {
    int k = k0 + ((v < 4) ? (kaHalf + 2 * v) : (16 + kaHalf + 2 * (v - 4)));
    float2 xi = *(const float2*)(x + (size_t)ri * DEE + k);
    float2 xj = *(const float2*)(x + (size_t)rj * DEE + k);
    a.h[2 * v]     = (_Float16)fmaxf(xi.x * xj.x, 0.f);
    a.h[2 * v + 1] = (_Float16)fmaxf(xi.y * xj.y, 0.f);
  }
}

// B fragment: 32x16 f16 tile from LDS weight panel stored transposed [n][k].
// lane column N = n0+ln; low half lanes K-offset 0, high half 8; elems 8..15 at K+16.
__device__ __forceinline__ void load_b_lds(const _Float16* __restrict__ wT,
                                           int n, int k0, int kHalf, AFrag& b) {
  const _Float16* p = wT + (size_t)n * DEE + k0 + kHalf;
#pragma unroll
  for (int v = 0; v < 8; ++v) {
    int o = (v < 4) ? (2 * v) : (16 + 2 * (v - 4));
    b.h[2 * v]     = p[o];
    b.h[2 * v + 1] = p[o + 1];
  }
}

// ---------------------------------------------------------------------------
// Kernel 1: edge-attr branch.  loss_rec[e] = sum_n BCE(sigmoid(decode1), t)
// accumulated into pos segment sums (+ counts).
// ---------------------------------------------------------------------------
__global__ __launch_bounds__(256) void attr_kernel(
    const float* __restrict__ x, const float* __restrict__ edge_attr,
    const float* __restrict__ W1, const float* __restrict__ b1,
    const float* __restrict__ W2, const float* __restrict__ b2,
    const int* __restrict__ edge_index, const int* __restrict__ edge_batch,
    float* __restrict__ seg_sum, float* __restrict__ seg_cnt, int E) {
  __shared__ _Float16 w1t[DEE * DEE];  // 32 KB, W1 transposed [n][k] f16
  for (int idx = threadIdx.x; idx < DEE * DEE; idx += 256) {
    int k = idx >> 7, n = idx & (DEE - 1);
    w1t[n * DEE + k] = (_Float16)W1[idx];
  }
  __syncthreads();

  const int wave = threadIdx.x >> 5;
  const int lane = threadIdx.x & 31;
  const int ln   = lane & 15;
  const int hi   = lane >> 4;
  const long long e0 = ((long long)blockIdx.x * 8 + wave) * 16;
  if (e0 >= E) return;  // wave-uniform: EXEC stays full for WMMA

  long long eM = e0 + ln; if (eM >= E) eM = E - 1;
  const int ri = edge_index[eM];
  const int rj = edge_index[(long long)E + eM];

  AFrag a[4];
#pragma unroll
  for (int t = 0; t < 4; ++t) load_a_em(x, ri, rj, t * 32, hi ? 8 : 0, a[t]);

  const int kHalf = hi ? 8 : 0;
  CFrag hc[8];  // H1 = relu(A@W1 + b1), 16x128 in C layout
#pragma unroll
  for (int nt = 0; nt < 8; ++nt) {
    float bias = b1[nt * 16 + ln];
    CFrag c;
#pragma unroll
    for (int i = 0; i < 8; ++i) c.f[i] = bias;
#pragma unroll
    for (int kt = 0; kt < 4; ++kt) {
      AFrag b; load_b_lds(w1t, nt * 16 + ln, kt * 32, kHalf, b);
      c.v = __builtin_amdgcn_wmma_f32_16x16x32_f16(false, a[kt].v, false, b.v,
                                                   (short)0, c.v, false, false);
    }
#pragma unroll
    for (int i = 0; i < 8; ++i) hc[nt].f[i] = fmaxf(c.f[i], 0.f);
  }

  // Per-lane slice of W2 (row-major [k][7]) for the D->7 projection.
  float w2r[8][7];
#pragma unroll
  for (int t = 0; t < 8; ++t)
#pragma unroll
    for (int n = 0; n < 7; ++n)
      w2r[t][n] = W2[(size_t)(t * 16 + ln) * 7 + n];
  float b2r = (ln < 7) ? b2[ln] : 0.f;

#pragma unroll
  for (int r = 0; r < 8; ++r) {
    const long long e = e0 + r + 8 * hi;
    float zk = 0.f;
#pragma unroll
    for (int n = 0; n < 7; ++n) {
      float z = 0.f;
#pragma unroll
      for (int t = 0; t < 8; ++t) z = fmaf(hc[t].f[r], w2r[t][n], z);
      z += __shfl_xor(z, 1, 16);
      z += __shfl_xor(z, 2, 16);
      z += __shfl_xor(z, 4, 16);
      z += __shfl_xor(z, 8, 16);
      if (ln == n) zk = z;  // lane ln<7 keeps logit for output n=ln
    }
    float bce = 0.f;
    if (ln < 7 && e < E) {
      float z  = zk + b2r;
      float tt = edge_attr[e * 9 + ln];
      bce = tt * sp_clamp(-z) + (1.f - tt) * sp_clamp(z);
    }
    bce += __shfl_xor(bce, 1, 16);
    bce += __shfl_xor(bce, 2, 16);
    bce += __shfl_xor(bce, 4, 16);
    bce += __shfl_xor(bce, 8, 16);
    if (ln == 0 && e < E) {
      int seg = edge_batch[e];
      atomicAdd(seg_sum + seg, bce);
      atomicAdd(seg_cnt + seg, 1.f);
    }
  }
}

// ---------------------------------------------------------------------------
// Kernel 2: edge-probability branch (shared We1/We2 for pos AND neg edges).
// tile < tiles -> positive edges (adds softplus(-z) into pos segment sums,
// count already done by kernel 1); else negative edges (softplus(z), + count).
// ---------------------------------------------------------------------------
__global__ __launch_bounds__(256) void edgeprob_kernel(
    const float* __restrict__ x,
    const float* __restrict__ We1, const float* __restrict__ be1,
    const float* __restrict__ We2, const float* __restrict__ be2,
    const int* __restrict__ ei_pos, const int* __restrict__ eb_pos,
    const int* __restrict__ ei_neg, const int* __restrict__ eb_neg,
    float* __restrict__ pos_sum,
    float* __restrict__ neg_sum, float* __restrict__ neg_cnt,
    float* __restrict__ aux, int E, int tiles) {
  __shared__ _Float16 wet[DEE * DEE];  // 32 KB, We1 transposed f16
  for (int idx = threadIdx.x; idx < DEE * DEE; idx += 256) {
    int k = idx >> 7, n = idx & (DEE - 1);
    wet[n * DEE + k] = (_Float16)We1[idx];
  }
  __syncthreads();

  const int wave = threadIdx.x >> 5;
  const int lane = threadIdx.x & 31;
  const int ln   = lane & 15;
  const int hi   = lane >> 4;
  int tile = blockIdx.x * 8 + wave;
  if (tile >= 2 * tiles) return;       // wave-uniform
  const bool is_pos = tile < tiles;    // wave-uniform
  if (!is_pos) tile -= tiles;
  const long long e0 = (long long)tile * 16;
  if (e0 >= E) return;

  const int* ei = is_pos ? ei_pos : ei_neg;
  long long eM = e0 + ln; if (eM >= E) eM = E - 1;
  const int ri = ei[eM];
  const int rj = ei[(long long)E + eM];

  AFrag a[4];
#pragma unroll
  for (int t = 0; t < 4; ++t) load_a_em(x, ri, rj, t * 32, hi ? 8 : 0, a[t]);

  const int kHalf = hi ? 8 : 0;
  CFrag hc[8];
#pragma unroll
  for (int nt = 0; nt < 8; ++nt) {
    float bias = be1[nt * 16 + ln];
    CFrag c;
#pragma unroll
    for (int i = 0; i < 8; ++i) c.f[i] = bias;
#pragma unroll
    for (int kt = 0; kt < 4; ++kt) {
      AFrag b; load_b_lds(wet, nt * 16 + ln, kt * 32, kHalf, b);
      c.v = __builtin_amdgcn_wmma_f32_16x16x32_f16(false, a[kt].v, false, b.v,
                                                   (short)0, c.v, false, false);
    }
#pragma unroll
    for (int i = 0; i < 8; ++i) hc[nt].f[i] = fmaxf(c.f[i], 0.f);
  }

  float we2r[8];
#pragma unroll
  for (int t = 0; t < 8; ++t) we2r[t] = We2[t * 16 + ln];
  const float be2v = be2[0];

  const int* eb = is_pos ? eb_pos : eb_neg;
  float auxacc = 0.f;
#pragma unroll
  for (int r = 0; r < 8; ++r) {
    float z = 0.f;
#pragma unroll
    for (int t = 0; t < 8; ++t) z = fmaf(hc[t].f[r], we2r[t], z);
    z += __shfl_xor(z, 1, 16);
    z += __shfl_xor(z, 2, 16);
    z += __shfl_xor(z, 4, 16);
    z += __shfl_xor(z, 8, 16);
    z += be2v;
    const float l = is_pos ? sp_clamp(-z) : sp_clamp(z);
    const long long e = e0 + r + 8 * hi;
    if (ln == 0 && e < E) {
      int seg = eb[e];
      if (is_pos) {
        atomicAdd(pos_sum + seg, l);
      } else {
        atomicAdd(neg_sum + seg, l);
        atomicAdd(neg_cnt + seg, 1.f);
      }
      auxacc += l;
    }
  }
  if (ln == 0) atomicAdd(aux + (is_pos ? 0 : 1), auxacc);
}

// ---------------------------------------------------------------------------
// Kernel 3: per-node KL, one wave per node (float4 loads, butterfly reduce).
// ---------------------------------------------------------------------------
__global__ __launch_bounds__(256) void kl_kernel(
    const float* __restrict__ xm, const float* __restrict__ xs,
    const int* __restrict__ batch,
    float* __restrict__ kl_sum, float* __restrict__ node_cnt, int N) {
  const int wave = threadIdx.x >> 5;
  const int lane = threadIdx.x & 31;
  const int n = blockIdx.x * 8 + wave;
  if (n >= N) return;
  float4 m = ((const float4*)(xm + (size_t)n * DEE))[lane];
  float4 s = ((const float4*)(xs + (size_t)n * DEE))[lane];
  float acc = 0.f;
  acc += 1.f + 2.f * logf(s.x) - m.x * m.x - s.x * s.x;
  acc += 1.f + 2.f * logf(s.y) - m.y * m.y - s.y * s.y;
  acc += 1.f + 2.f * logf(s.z) - m.z * m.z - s.z * s.z;
  acc += 1.f + 2.f * logf(s.w) - m.w * m.w - s.w * s.w;
  acc += __shfl_xor(acc, 1, 32);
  acc += __shfl_xor(acc, 2, 32);
  acc += __shfl_xor(acc, 4, 32);
  acc += __shfl_xor(acc, 8, 32);
  acc += __shfl_xor(acc, 16, 32);
  if (lane == 0) {
    int g = batch[n];
    atomicAdd(kl_sum + g, -0.5f * acc);
    atomicAdd(node_cnt + g, 1.f);
  }
}

// ---------------------------------------------------------------------------
// Kernel 4: finalize -> (loss, aux).
// ---------------------------------------------------------------------------
__global__ __launch_bounds__(1024) void finalize_kernel(
    const float* __restrict__ pos_sum, const float* __restrict__ pos_cnt,
    const float* __restrict__ neg_sum, const float* __restrict__ neg_cnt,
    const float* __restrict__ kl_sum, const float* __restrict__ node_cnt,
    const float* __restrict__ aux, const int* __restrict__ n_graphs,
    float* __restrict__ out, int E) {
  __shared__ float red[32];
  const int G = *n_graphs;
  float v = 0.f;
  for (int g = threadIdx.x; g < G; g += blockDim.x) {
    float pc = fmaxf(pos_cnt[g], 1.f);
    float nc = fmaxf(neg_cnt[g], 1.f);
    float kc = fmaxf(node_cnt[g], 1.f);
    v += pos_sum[g] / pc + neg_sum[g] / nc + kl_sum[g] / (kc * kc);
  }
  v += __shfl_xor(v, 1, 32);
  v += __shfl_xor(v, 2, 32);
  v += __shfl_xor(v, 4, 32);
  v += __shfl_xor(v, 8, 32);
  v += __shfl_xor(v, 16, 32);
  const int lane = threadIdx.x & 31, wv = threadIdx.x >> 5;
  if (lane == 0) red[wv] = v;
  __syncthreads();
  if (threadIdx.x < 32) {
    float s = (threadIdx.x < 32) ? red[threadIdx.x] : 0.f;
    s += __shfl_xor(s, 1, 32);
    s += __shfl_xor(s, 2, 32);
    s += __shfl_xor(s, 4, 32);
    s += __shfl_xor(s, 8, 32);
    s += __shfl_xor(s, 16, 32);
    if (threadIdx.x == 0) {
      out[0] = s / (float)G;
      out[1] = 0.5f * (aux[0] + aux[1]) / (float)E;
    }
  }
}

extern "C" void kernel_launch(void* const* d_in, const int* in_sizes, int n_in,
                              void* d_out, int out_size, void* d_ws, size_t ws_size,
                              hipStream_t stream) {
  const float* x         = (const float*)d_in[0];
  const float* edge_attr = (const float*)d_in[1];
  const float* x_mean    = (const float*)d_in[2];
  const float* x_std     = (const float*)d_in[3];
  const float* W1        = (const float*)d_in[4];
  const float* b1        = (const float*)d_in[5];
  const float* W2        = (const float*)d_in[6];
  const float* b2        = (const float*)d_in[7];
  const float* We1       = (const float*)d_in[8];
  const float* be1       = (const float*)d_in[9];
  const float* We2       = (const float*)d_in[10];
  const float* be2       = (const float*)d_in[11];
  const int* edge_index      = (const int*)d_in[12];
  const int* edge_index_neg  = (const int*)d_in[13];
  const int* eib             = (const int*)d_in[14];
  const int* einb            = (const int*)d_in[15];
  const int* batch           = (const int*)d_in[16];
  const int* n_graphs        = (const int*)d_in[17];

  const int E = in_sizes[12] / 2;
  const int N = in_sizes[0] / DEE;

  const int GMAX = 2048;  // setup uses G=1024
  float* ws       = (float*)d_ws;
  float* pos_sum  = ws;
  float* pos_cnt  = ws + 1 * GMAX;
  float* neg_sum  = ws + 2 * GMAX;
  float* neg_cnt  = ws + 3 * GMAX;
  float* kl_sum   = ws + 4 * GMAX;
  float* node_cnt = ws + 5 * GMAX;
  float* aux      = ws + 6 * GMAX;  // 2 floats

  hipMemsetAsync(d_ws, 0, (size_t)(6 * GMAX + 2) * sizeof(float), stream);

  const int tiles = (E + 15) / 16;
  const int blocks1 = (tiles + 7) / 8;
  attr_kernel<<<blocks1, 256, 0, stream>>>(x, edge_attr, W1, b1, W2, b2,
                                           edge_index, eib, pos_sum, pos_cnt, E);
  const int blocks2 = (2 * tiles + 7) / 8;
  edgeprob_kernel<<<blocks2, 256, 0, stream>>>(x, We1, be1, We2, be2,
                                               edge_index, eib, edge_index_neg, einb,
                                               pos_sum, neg_sum, neg_cnt, aux, E, tiles);
  const int blocks3 = (N + 7) / 8;
  kl_kernel<<<blocks3, 256, 0, stream>>>(x_mean, x_std, batch, kl_sum, node_cnt, N);
  finalize_kernel<<<1, 1024, 0, stream>>>(pos_sum, pos_cnt, neg_sum, neg_cnt,
                                          kl_sum, node_cnt, aux, n_graphs,
                                          (float*)d_out, E);
}
